// pointSem_84293028152076
// MI455X (gfx1250) — compile-verified, matrix operand order checked by reading, not run on hardware
//
#include <hip/hip_runtime.h>
#include <hip/hip_bf16.h>

#define Bb   16
#define Nn   1280
#define NCLS 16

typedef __attribute__((ext_vector_type(16))) _Float16 v16h;
typedef __attribute__((ext_vector_type(8)))  _Float16 v8h;
typedef __attribute__((ext_vector_type(8)))  float    v8f;

// ---------------- WMMA fragment helpers (v_wmma_f32_16x16x32_f16, wave32) ----------------
// 16x32 f16 fragment from a row-major, 16B-aligned f16 matrix (LDS or global), stride = Kpad
// (multiple of 32). Lane l -> row l&15; K window split per ISA layout: lanes<16 K{0..7,16..23},
// lanes>=16 K{8..15,24..31}. Two contiguous 16-byte loads per lane.
__device__ __forceinline__ v16h frag_h(const _Float16* __restrict__ base, int stride, int lane) {
  const _Float16* p = base + (size_t)(lane & 15) * stride + ((lane >> 4) << 3);
  v8h lo = *(const v8h*)(p);
  v8h hi = *(const v8h*)(p + 16);
  v16h f;
#pragma unroll
  for (int j = 0; j < 8; ++j) { f[j] = lo[j]; f[8 + j] = hi[j]; }
  return f;
}

// 16x32 fragment from row-major fp32 matrix with ld a multiple of 32 (no bounds needed).
// Four aligned float4 loads + cvt per lane.
__device__ __forceinline__ v16h frag_a_f32(const float* __restrict__ A, int ld, int r0, int k0, int lane) {
  const float* p = A + (size_t)(r0 + (lane & 15)) * ld + (k0 + ((lane >> 4) << 3));
  float4 a0 = *(const float4*)(p);
  float4 a1 = *(const float4*)(p + 4);
  float4 a2 = *(const float4*)(p + 16);
  float4 a3 = *(const float4*)(p + 20);
  v16h f;
  f[0]  = (_Float16)a0.x; f[1]  = (_Float16)a0.y; f[2]  = (_Float16)a0.z; f[3]  = (_Float16)a0.w;
  f[4]  = (_Float16)a1.x; f[5]  = (_Float16)a1.y; f[6]  = (_Float16)a1.z; f[7]  = (_Float16)a1.w;
  f[8]  = (_Float16)a2.x; f[9]  = (_Float16)a2.y; f[10] = (_Float16)a2.z; f[11] = (_Float16)a2.w;
  f[12] = (_Float16)a3.x; f[13] = (_Float16)a3.y; f[14] = (_Float16)a3.z; f[15] = (_Float16)a3.w;
  return f;
}

// ---------------- weight conversion: fp32 (N,K) -> f16 (N,Kpad) zero-padded ----------------
__global__ __launch_bounds__(256) void convert_w_kernel(const float* __restrict__ W,
                                                        _Float16* __restrict__ Wh,
                                                        int N, int K, int Kpad) {
  size_t e = (size_t)blockIdx.x * blockDim.x + threadIdx.x;
  if (e >= (size_t)N * Kpad) return;
  int k = (int)(e % Kpad);
  int n = (int)(e / Kpad);
  Wh[e] = (k < K) ? (_Float16)W[(size_t)n * K + k] : (_Float16)0.f;
}

// ---------------- layout / prep kernels ----------------
__global__ __launch_bounds__(256) void transpose_kernel(const float* __restrict__ xin,
                                                        float* __restrict__ pts0,
                                                        float* __restrict__ xyz0) {
  size_t e = (size_t)blockIdx.x * blockDim.x + threadIdx.x;
  size_t total = (size_t)Bb * Nn * 9;
  if (e >= total) return;
  int c = (int)(e % 9);
  size_t q = e / 9;                 // b*Nn + n
  int b = (int)(q / Nn), n = (int)(q % Nn);
  float v = xin[((size_t)b * 9 + c) * Nn + n];
  pts0[e] = v;
  if (c < 3) xyz0[q * 3 + c] = v;
}

// Farthest point sampling: one block per batch. Writes new_xyz directly (gathered).
__global__ __launch_bounds__(256) void fps_kernel(const float* __restrict__ xyz, int Nin, int npoint,
                                                  float* __restrict__ new_xyz) {
  __shared__ float dist[1280];
  __shared__ float rv[256];
  __shared__ int   ri[256];
  __shared__ int   sidx[1024];
  __shared__ int   cur;
  int b = blockIdx.x, tid = threadIdx.x;
  const float* X = xyz + (size_t)b * Nin * 3;
  for (int n = tid; n < Nin; n += 256) dist[n] = 1e10f;
  if (tid == 0) cur = 0;
  __syncthreads();
  for (int it = 0; it < npoint; ++it) {
    int far = cur;
    if (tid == 0) sidx[it] = far;
    float cx = X[far * 3 + 0], cy = X[far * 3 + 1], cz = X[far * 3 + 2];
    float bestv = -1.f; int besti = 0;
    for (int n = tid; n < Nin; n += 256) {
      float dx = X[n * 3 + 0] - cx, dy = X[n * 3 + 1] - cy, dz = X[n * 3 + 2] - cz;
      float d = fminf(dist[n], dx * dx + dy * dy + dz * dz);
      dist[n] = d;
      if (d > bestv) { bestv = d; besti = n; }
    }
    rv[tid] = bestv; ri[tid] = besti;
    __syncthreads();
    for (int s = 128; s > 0; s >>= 1) {
      if (tid < s) {
        if (rv[tid + s] > rv[tid] || (rv[tid + s] == rv[tid] && ri[tid + s] < ri[tid])) {
          rv[tid] = rv[tid + s]; ri[tid] = ri[tid + s];
        }
      }
      __syncthreads();
    }
    if (tid == 0) cur = ri[0];
    __syncthreads();
  }
  for (int e = tid; e < npoint * 3; e += 256) {
    int p = e / 3, d = e % 3;
    new_xyz[((size_t)b * npoint + p) * 3 + d] = X[sidx[p] * 3 + d];
  }
}

// Ball query: one thread per query point; indices in ascending order (matches sorted gi).
__global__ __launch_bounds__(256) void ball_query_kernel(const float* __restrict__ xyz,
                                                         const float* __restrict__ new_xyz,
                                                         int Nin, int npoint, float r2,
                                                         int* __restrict__ gidx) {
  int q = blockIdx.x * blockDim.x + threadIdx.x;
  if (q >= Bb * npoint) return;
  int b = q / npoint;
  const float* X = xyz + (size_t)b * Nin * 3;
  float cx = new_xyz[(size_t)q * 3 + 0], cy = new_xyz[(size_t)q * 3 + 1], cz = new_xyz[(size_t)q * 3 + 2];
  int* out = gidx + (size_t)q * 32;
  int cnt = 0;
  for (int n = 0; n < Nin && cnt < 32; ++n) {
    float dx = X[n * 3 + 0] - cx, dy = X[n * 3 + 1] - cy, dz = X[n * 3 + 2] - cz;
    if (dx * dx + dy * dy + dz * dz <= r2) out[cnt++] = n;
  }
  int first = (cnt > 0) ? out[0] : (Nin - 1);
  for (int k = cnt; k < 32; ++k) out[k] = first;
}

// ---------------- fused SA MLP stack: group in LDS, 3 WMMA layers, max-pool ----------------
__device__ __forceinline__ void lds_mlp_layer(const _Float16* __restrict__ in, int Kpad,
                                              const _Float16* __restrict__ Wh,
                                              const float* __restrict__ bias,
                                              const float* __restrict__ gain,
                                              const float* __restrict__ beta,
                                              _Float16* __restrict__ out, int Cout, int tid) {
  int wave = tid >> 5, lane = tid & 31;
  int ntiles = (Cout >> 4) << 1;           // 2 M-tiles (32 rows) x Cout/16 N-tiles
  for (int t = wave; t < ntiles; t += 8) {
    int m0 = (t & 1) << 4;
    int n0 = (t >> 1) << 4;
    v8f acc = {};
    for (int k0 = 0; k0 < Kpad; k0 += 32) {
      v16h a = frag_h(in + (size_t)m0 * Kpad + k0, Kpad, lane);
      v16h w = frag_h(Wh + (size_t)n0 * Kpad + k0, Kpad, lane);
      acc = __builtin_amdgcn_wmma_f32_16x16x32_f16(false, a, false, w, (short)0, acc, false, false);
    }
    int n  = n0 + (lane & 15);
    int mb = m0 + ((lane >> 4) << 3);
    float bb = bias[n], gg = gain[n], ee = beta[n];
#pragma unroll
    for (int r = 0; r < 8; ++r) {
      float v = (acc[r] + bb) * gg + ee;
      out[(size_t)(mb + r) * Cout + n] = (_Float16)(v > 0.f ? v : 0.f);
    }
  }
}

__global__ __launch_bounds__(256) void sa_fused_kernel(
    const float* __restrict__ xyz, const float* __restrict__ pts,
    const int* __restrict__ gidx, const float* __restrict__ new_xyz,
    const _Float16* W0, const float* B0, const float* G0, const float* E0,
    const _Float16* W1, const float* B1, const float* G1, const float* E1,
    const _Float16* W2, const float* B2, const float* G2, const float* E2,
    float* __restrict__ outp, int Nin, int Cpts, int npoint, int c1, int c2, int c3) {
  __shared__ _Float16 bufA[32 * 288];     // input (cin<=259 pad 288) and layer-2 out (<=256)
  __shared__ _Float16 bufB[32 * 512];     // layer-1 out (<=256) and layer-3 out (<=512)
  __shared__ int   sg[32];
  __shared__ float ctr[3];
  int grp = blockIdx.x;                   // b*npoint + p
  int b = grp / npoint;
  int tid = threadIdx.x;
  int cin = 3 + Cpts;
  int K0 = (cin + 31) & ~31;
  if (tid < 32) sg[tid] = gidx[(size_t)grp * 32 + tid];
  if (tid < 3)  ctr[tid] = new_xyz[(size_t)grp * 3 + tid];
  __syncthreads();
  const float* Xb = xyz + (size_t)b * Nin * 3;
  const float* Pb = pts + (size_t)b * Nin * Cpts;
  for (int e = tid; e < 32 * K0; e += 256) {
    int s = e / K0, k = e - s * K0;
    float v = 0.f;
    int n = sg[s];
    if (k < 3)        v = Xb[(size_t)n * 3 + k] - ctr[k];
    else if (k < cin) v = Pb[(size_t)n * Cpts + (k - 3)];
    bufA[(size_t)s * K0 + k] = (_Float16)v;
  }
  __syncthreads();
  lds_mlp_layer(bufA, K0, W0, B0, G0, E0, bufB, c1, tid);
  __syncthreads();
  lds_mlp_layer(bufB, c1, W1, B1, G1, E1, bufA, c2, tid);
  __syncthreads();
  lds_mlp_layer(bufA, c2, W2, B2, G2, E2, bufB, c3, tid);
  __syncthreads();
  for (int c = tid; c < c3; c += 256) {
    float m = -1e30f;
    for (int s = 0; s < 32; ++s) m = fmaxf(m, (float)bufB[(size_t)s * c3 + c]);
    outp[(size_t)grp * c3 + c] = m;
  }
}

// ---------------- generic WMMA GEMM + BN/ReLU (fp32 A/out, f16 W; K % 32 == 0) ----------------
__global__ __launch_bounds__(256) void gemm_bn_kernel(
    const float* __restrict__ A, const _Float16* __restrict__ Wh,
    const float* __restrict__ bias, const float* __restrict__ gain, const float* __restrict__ beta,
    float* __restrict__ out, int M, int K, int N, int mode) {
  int wave = (int)(threadIdx.x >> 5), lane = (int)(threadIdx.x & 31);
  int tilesN = N >> 4;
  int tile = blockIdx.x * 8 + wave;
  if (tile >= (M >> 5) * tilesN) return;          // wave-uniform: EXEC stays full for WMMA
  int m0 = (tile / tilesN) << 5;                  // 32-row M tile, W fragment reused twice
  int n0 = (tile % tilesN) << 4;
  v8f acc0 = {}, acc1 = {};
  for (int k0 = 0; k0 < K; k0 += 32) {
    v16h w  = frag_h(Wh + (size_t)n0 * K + k0, K, lane);
    v16h a0 = frag_a_f32(A, K, m0,      k0, lane);
    v16h a1 = frag_a_f32(A, K, m0 + 16, k0, lane);
    acc0 = __builtin_amdgcn_wmma_f32_16x16x32_f16(false, a0, false, w, (short)0, acc0, false, false);
    acc1 = __builtin_amdgcn_wmma_f32_16x16x32_f16(false, a1, false, w, (short)0, acc1, false, false);
  }
  int n  = n0 + (lane & 15);
  int mb = m0 + ((lane >> 4) << 3);
  float bb = bias[n];
  if (mode) {
    float gg = gain[n], ee = beta[n];
#pragma unroll
    for (int r = 0; r < 8; ++r) {
      float v0 = (acc0[r] + bb) * gg + ee;
      float v1 = (acc1[r] + bb) * gg + ee;
      out[(size_t)(mb + r) * N + n]      = v0 > 0.f ? v0 : 0.f;
      out[(size_t)(mb + 16 + r) * N + n] = v1 > 0.f ? v1 : 0.f;
    }
  } else {
#pragma unroll
    for (int r = 0; r < 8; ++r) {
      out[(size_t)(mb + r) * N + n]      = acc0[r] + bb;
      out[(size_t)(mb + 16 + r) * N + n] = acc1[r] + bb;
    }
  }
}

// ---------------- feature propagation: 3-NN + weighted interp/concat ----------------
__global__ __launch_bounds__(256) void knn3_kernel(const float* __restrict__ xyz1,
                                                   const float* __restrict__ xyz2,
                                                   int N1, int N2,
                                                   int* __restrict__ kidx, float* __restrict__ kw) {
  int q = blockIdx.x * blockDim.x + threadIdx.x;
  if (q >= Bb * N1) return;
  int b = q / N1;
  const float* X2 = xyz2 + (size_t)b * N2 * 3;
  float px = xyz1[(size_t)q * 3 + 0], py = xyz1[(size_t)q * 3 + 1], pz = xyz1[(size_t)q * 3 + 2];
  float d0 = 1e30f, d1 = 1e30f, d2 = 1e30f; int i0 = 0, i1 = 0, i2 = 0;
  for (int n = 0; n < N2; ++n) {
    float dx = X2[n * 3 + 0] - px, dy = X2[n * 3 + 1] - py, dz = X2[n * 3 + 2] - pz;
    float d = dx * dx + dy * dy + dz * dz;
    if (d < d0)      { d2 = d1; i2 = i1; d1 = d0; i1 = i0; d0 = d; i0 = n; }
    else if (d < d1) { d2 = d1; i2 = i1; d1 = d; i1 = n; }
    else if (d < d2) { d2 = d; i2 = n; }
  }
  float r0 = 1.f / (d0 + 1e-8f), r1 = 1.f / (d1 + 1e-8f), r2 = 1.f / (d2 + 1e-8f);
  float s = r0 + r1 + r2;
  kidx[(size_t)q * 3 + 0] = i0; kidx[(size_t)q * 3 + 1] = i1; kidx[(size_t)q * 3 + 2] = i2;
  kw[(size_t)q * 3 + 0] = r0 / s; kw[(size_t)q * 3 + 1] = r1 / s; kw[(size_t)q * 3 + 2] = r2 / s;
}

__global__ __launch_bounds__(256) void fp_concat_kernel(const float* __restrict__ pts1, int C1,
                                                        const float* __restrict__ pts2, int C2, int N2,
                                                        const int* __restrict__ kidx,
                                                        const float* __restrict__ kw,
                                                        int N1, float* __restrict__ H) {
  int Ct = C1 + C2;
  size_t e = (size_t)blockIdx.x * blockDim.x + threadIdx.x;
  if (e >= (size_t)Bb * N1 * Ct) return;
  int c = (int)(e % Ct);
  size_t q = e / Ct;
  int b = (int)(q / N1);
  float v;
  if (c < C1) {
    v = pts1[q * C1 + c];
  } else {
    int cc = c - C1;
    v = 0.f;
#pragma unroll
    for (int j = 0; j < 3; ++j)
      v += kw[q * 3 + j] * pts2[((size_t)b * N2 + kidx[q * 3 + j]) * C2 + cc];
  }
  H[e] = v;
}

__global__ __launch_bounds__(256) void log_softmax_kernel(float* __restrict__ logits) {
  int q = blockIdx.x * blockDim.x + threadIdx.x;
  if (q >= Bb * Nn) return;
  float* p = logits + (size_t)q * NCLS;
  float mx = p[0];
#pragma unroll
  for (int c = 1; c < NCLS; ++c) mx = fmaxf(mx, p[c]);
  float s = 0.f;
#pragma unroll
  for (int c = 0; c < NCLS; ++c) s += expf(p[c] - mx);
  float l = logf(s) + mx;
#pragma unroll
  for (int c = 0; c < NCLS; ++c) p[c] = p[c] - l;
}

// ---------------- host side ----------------
struct ParamMap {
  const float *saW[4][3], *saB[4][3], *saG[4][3], *saE[4][3];
  const float *fpW[4][3], *fpB[4][3], *fpG[4][3], *fpE[4][3];
  const float *hW1, *hB1, *hG1, *hE1, *hW2, *hB2;
  const float *xyz;
};

static void build_map(void* const* din, const int* sizes, ParamMap* m) {
  const int fpCum[4] = {0, 2, 4, 6};
  const int fpL[4] = {2, 2, 2, 3};
  bool insertion = (sizes[0] == Bb * 9 * Nn);   // xyz first -> insertion-order flattening
  if (insertion) {
    m->xyz = (const float*)din[0];
    for (int s = 0; s < 4; ++s)
      for (int l = 0; l < 3; ++l) {
        int base = 1 + (s * 3 + l) * 4;                       // W, b, g, be
        m->saW[s][l] = (const float*)din[base + 0];
        m->saB[s][l] = (const float*)din[base + 1];
        m->saG[s][l] = (const float*)din[base + 2];
        m->saE[s][l] = (const float*)din[base + 3];
      }
    for (int s = 0; s < 4; ++s)
      for (int l = 0; l < fpL[s]; ++l) {
        int base = 49 + (fpCum[s] + l) * 4;                   // W, b, g, be
        m->fpW[s][l] = (const float*)din[base + 0];
        m->fpB[s][l] = (const float*)din[base + 1];
        m->fpG[s][l] = (const float*)din[base + 2];
        m->fpE[s][l] = (const float*)din[base + 3];
      }
    m->hW1 = (const float*)din[85]; m->hB1 = (const float*)din[86];
    m->hG1 = (const float*)din[87]; m->hE1 = (const float*)din[88];
    m->hW2 = (const float*)din[89]; m->hB2 = (const float*)din[90];
  } else {                                                    // jax sorted-key flattening
    for (int s = 0; s < 4; ++s)
      for (int l = 0; l < fpL[s]; ++l) {
        int base = (fpCum[s] + l) * 4;                        // W, b, be, g
        m->fpW[s][l] = (const float*)din[base + 0];
        m->fpB[s][l] = (const float*)din[base + 1];
        m->fpE[s][l] = (const float*)din[base + 2];
        m->fpG[s][l] = (const float*)din[base + 3];
      }
    m->hW1 = (const float*)din[36]; m->hW2 = (const float*)din[37];
    m->hB1 = (const float*)din[38]; m->hB2 = (const float*)din[39];
    m->hE1 = (const float*)din[40]; m->hG1 = (const float*)din[41];
    for (int s = 0; s < 4; ++s)
      for (int l = 0; l < 3; ++l) {
        int base = 42 + (s * 3 + l) * 4;                      // W, b, be, g
        m->saW[s][l] = (const float*)din[base + 0];
        m->saB[s][l] = (const float*)din[base + 1];
        m->saE[s][l] = (const float*)din[base + 2];
        m->saG[s][l] = (const float*)din[base + 3];
      }
    m->xyz = (const float*)din[90];
  }
}

static void launch_gemm(const float* A, const _Float16* Wh, const float* b, const float* g,
                        const float* e, float* out, int M, int K, int N, int mode, hipStream_t st) {
  int tiles = (M / 32) * (N / 16);
  gemm_bn_kernel<<<(tiles + 7) / 8, 256, 0, st>>>(A, Wh, b, g, e, out, M, K, N, mode);
}

extern "C" void kernel_launch(void* const* d_in, const int* in_sizes, int n_in,
                              void* d_out, int out_size, void* d_ws, size_t ws_size,
                              hipStream_t stream) {
  if (n_in < 91) return;
  ParamMap m;
  build_map(d_in, in_sizes, &m);

  // workspace bump allocator
  char* w = (char*)d_ws;
  auto alloc = [&](size_t bytes) -> void* {
    void* p = (void*)w;
    w += (bytes + 255) & ~(size_t)255;
    return p;
  };
  // f16 weight conversion (zero-padded K)
  auto convertW = [&](const float* W, int N, int K, int Kpad) -> _Float16* {
    _Float16* Wh = (_Float16*)alloc((size_t)N * Kpad * 2);
    size_t tot = (size_t)N * Kpad;
    convert_w_kernel<<<(unsigned)((tot + 255) / 256), 256, 0, stream>>>(W, Wh, N, K, Kpad);
    return Wh;
  };

  const int NsaIn[4]  = {1280, 1024, 256, 64};
  const int NsaOut[4] = {1024, 256, 64, 16};
  const int CptsA[4]  = {9, 64, 128, 256};
  const int C1s[4] = {32, 64, 128, 256}, C2s[4] = {32, 64, 128, 256}, C3s[4] = {64, 128, 256, 512};
  const float rad[4] = {0.1f, 0.2f, 0.4f, 0.8f};

  float* pts0 = (float*)alloc((size_t)Bb * Nn * 9 * 4);
  float* xyzL[5]; float* ptsL[5];
  xyzL[0] = (float*)alloc((size_t)Bb * Nn * 3 * 4);
  ptsL[0] = pts0;
  for (int s = 0; s < 4; ++s) {
    xyzL[s + 1] = (float*)alloc((size_t)Bb * NsaOut[s] * 3 * 4);
    ptsL[s + 1] = (float*)alloc((size_t)Bb * NsaOut[s] * C3s[s] * 4);
  }
  float* p3 = (float*)alloc((size_t)Bb * 64 * 256 * 4);
  float* p2 = (float*)alloc((size_t)Bb * 256 * 256 * 4);
  float* p1 = (float*)alloc((size_t)Bb * 1024 * 128 * 4);
  float* p0 = (float*)alloc((size_t)Bb * 1280 * 128 * 4);
  float* HA = (float*)alloc((size_t)Bb * 1024 * 320 * 4);   // max concat buffer
  float* HB = (float*)alloc((size_t)Bb * 1024 * 256 * 4);   // max hidden buffer
  int*   gIdx = (int*)alloc((size_t)Bb * 1024 * 32 * 4);
  int*   kIdx = (int*)alloc((size_t)Bb * 1280 * 3 * 4);
  float* kW   = (float*)alloc((size_t)Bb * 1280 * 3 * 4);
  float* outF = (float*)d_out;

  // convert all weights to padded f16 (cheap: ~2MB total)
  _Float16* saWh[4][3]; _Float16* fpWh[4][3];
  for (int s = 0; s < 4; ++s) {
    int K0 = (3 + CptsA[s] + 31) & ~31;
    saWh[s][0] = convertW(m.saW[s][0], C1s[s], 3 + CptsA[s], K0);
    saWh[s][1] = convertW(m.saW[s][1], C2s[s], C1s[s], C1s[s]);
    saWh[s][2] = convertW(m.saW[s][2], C3s[s], C2s[s], C2s[s]);
  }
  const int fpKa[4][3] = {{768, 256, 0}, {384, 256, 0}, {320, 256, 0}, {128, 128, 128}};
  const int fpNa[4][3] = {{256, 256, 0}, {256, 256, 0}, {256, 128, 0}, {128, 128, 128}};
  const int fpLn[4] = {2, 2, 2, 3};
  for (int s = 0; s < 4; ++s)
    for (int l = 0; l < fpLn[s]; ++l)
      fpWh[s][l] = convertW(m.fpW[s][l], fpNa[s][l], fpKa[s][l], fpKa[s][l]);
  _Float16* hW1h = convertW(m.hW1, 128, 128, 128);
  _Float16* hW2h = convertW(m.hW2, NCLS, 128, 128);

  // transpose (B,9,N) -> pts (B,N,9) and xyz (B,N,3)
  {
    size_t tot = (size_t)Bb * Nn * 9;
    transpose_kernel<<<(unsigned)((tot + 255) / 256), 256, 0, stream>>>(m.xyz, pts0, xyzL[0]);
  }

  // ---- set abstraction stages ----
  for (int s = 0; s < 4; ++s) {
    int Nin = NsaIn[s], np = NsaOut[s];
    fps_kernel<<<Bb, 256, 0, stream>>>(xyzL[s], Nin, np, xyzL[s + 1]);
    int tq = Bb * np;
    ball_query_kernel<<<(tq + 255) / 256, 256, 0, stream>>>(xyzL[s], xyzL[s + 1], Nin, np,
                                                            rad[s] * rad[s], gIdx);
    sa_fused_kernel<<<Bb * np, 256, 0, stream>>>(
        xyzL[s], ptsL[s], gIdx, xyzL[s + 1],
        saWh[s][0], m.saB[s][0], m.saG[s][0], m.saE[s][0],
        saWh[s][1], m.saB[s][1], m.saG[s][1], m.saE[s][1],
        saWh[s][2], m.saB[s][2], m.saG[s][2], m.saE[s][2],
        ptsL[s + 1], Nin, CptsA[s], np, C1s[s], C2s[s], C3s[s]);
  }

  // ---- feature propagation ----
  // FP0: (64 <- 16), concat 256+512=768 -> 256 -> 256  => p3
  knn3_kernel<<<(Bb * 64 + 255) / 256, 256, 0, stream>>>(xyzL[3], xyzL[4], 64, 16, kIdx, kW);
  { size_t tot = (size_t)Bb * 64 * 768;
    fp_concat_kernel<<<(unsigned)((tot + 255) / 256), 256, 0, stream>>>(ptsL[3], 256, ptsL[4], 512, 16, kIdx, kW, 64, HA); }
  launch_gemm(HA, fpWh[0][0], m.fpB[0][0], m.fpG[0][0], m.fpE[0][0], HB, Bb * 64, 768, 256, 1, stream);
  launch_gemm(HB, fpWh[0][1], m.fpB[0][1], m.fpG[0][1], m.fpE[0][1], p3, Bb * 64, 256, 256, 1, stream);

  // FP1: (256 <- 64), concat 128+256=384 -> 256 -> 256 => p2
  knn3_kernel<<<(Bb * 256 + 255) / 256, 256, 0, stream>>>(xyzL[2], xyzL[3], 256, 64, kIdx, kW);
  { size_t tot = (size_t)Bb * 256 * 384;
    fp_concat_kernel<<<(unsigned)((tot + 255) / 256), 256, 0, stream>>>(ptsL[2], 128, p3, 256, 64, kIdx, kW, 256, HA); }
  launch_gemm(HA, fpWh[1][0], m.fpB[1][0], m.fpG[1][0], m.fpE[1][0], HB, Bb * 256, 384, 256, 1, stream);
  launch_gemm(HB, fpWh[1][1], m.fpB[1][1], m.fpG[1][1], m.fpE[1][1], p2, Bb * 256, 256, 256, 1, stream);

  // FP2: (1024 <- 256), concat 64+256=320 -> 256 -> 128 => p1
  knn3_kernel<<<(Bb * 1024 + 255) / 256, 256, 0, stream>>>(xyzL[1], xyzL[2], 1024, 256, kIdx, kW);
  { size_t tot = (size_t)Bb * 1024 * 320;
    fp_concat_kernel<<<(unsigned)((tot + 255) / 256), 256, 0, stream>>>(ptsL[1], 64, p2, 256, 256, kIdx, kW, 1024, HA); }
  launch_gemm(HA, fpWh[2][0], m.fpB[2][0], m.fpG[2][0], m.fpE[2][0], HB, Bb * 1024, 320, 256, 1, stream);
  launch_gemm(HB, fpWh[2][1], m.fpB[2][1], m.fpG[2][1], m.fpE[2][1], p1, Bb * 1024, 256, 128, 1, stream);

  // FP3: (1280 <- 1024), interp-only 128 -> 128 -> 128 -> 128 => p0
  knn3_kernel<<<(Bb * 1280 + 255) / 256, 256, 0, stream>>>(xyzL[0], xyzL[1], 1280, 1024, kIdx, kW);
  { size_t tot = (size_t)Bb * 1280 * 128;
    fp_concat_kernel<<<(unsigned)((tot + 255) / 256), 256, 0, stream>>>(nullptr, 0, p1, 128, 1024, kIdx, kW, 1280, HA); }
  launch_gemm(HA, fpWh[3][0], m.fpB[3][0], m.fpG[3][0], m.fpE[3][0], HB, Bb * 1280, 128, 128, 1, stream);
  launch_gemm(HB, fpWh[3][1], m.fpB[3][1], m.fpG[3][1], m.fpE[3][1], HA, Bb * 1280, 128, 128, 1, stream);
  launch_gemm(HA, fpWh[3][2], m.fpB[3][2], m.fpG[3][2], m.fpE[3][2], p0, Bb * 1280, 128, 128, 1, stream);

  // ---- head ----
  launch_gemm(p0, hW1h, m.hB1, m.hG1, m.hE1, HB, Bb * 1280, 128, 128, 1, stream);
  launch_gemm(HB, hW2h, m.hB2, m.hB2, m.hB2, outF, Bb * 1280, 128, NCLS, 0, stream);
  log_softmax_kernel<<<(Bb * Nn + 255) / 256, 256, 0, stream>>>(outF);

  (void)out_size; (void)ws_size;
}
